// RelativePositionEnhancedSA_63977832841573
// MI455X (gfx1250) — matrix operand order repaired and checked by reading, not compiled, not run on hardware
//
#include <hip/hip_runtime.h>
#include <hip/hip_bf16.h>

typedef __attribute__((ext_vector_type(16))) _Float16 v16h;
typedef __attribute__((ext_vector_type(8)))  float    v8f;

union AFrag { v16h v; _Float16 h[16]; unsigned u[8]; };
union BFrag { v16h v; _Float16 h[16]; unsigned u[8]; };
union CFrag { v8f  v; float f[8]; };

#define BM 128
#define BN 128
#define BK 32

// ---------------------------------------------------------------------------
// f32 -> f16 conversion (elementwise)
// ---------------------------------------------------------------------------
__global__ __launch_bounds__(256) void cvt_f32_f16(const float* __restrict__ in,
                                                   _Float16* __restrict__ out, int n) {
    int i = blockIdx.x * 256 + threadIdx.x;
    if (i < n) out[i] = (_Float16)in[i];
}

// ---------------------------------------------------------------------------
// GEMM: Out[m,n] = sum_k A[m,k] * W[n,k] + bias[n]
// A: M x K f16 row-major, W: Nout x K f16 row-major.
// Double-buffered LDS staging via GLOBAL_LOAD_ASYNC_TO_LDS_B128 (ASYNCcnt),
// overlapped with v_wmma_f32_16x16x32_f16 compute.
// Grid: (Nout/128, M/128), Block: 256 (8 wave32s as 2(M) x 4(N)).
// ---------------------------------------------------------------------------
__global__ __launch_bounds__(256) void gemm_f16(
    const _Float16* __restrict__ A, const _Float16* __restrict__ W,
    const float* __restrict__ bias, float* outF32, _Float16* outF16,
    int Nout, int K, int accFlag)
{
    __shared__ __align__(16) _Float16 As[2][128][40];
    __shared__ __align__(16) _Float16 Bs[2][128][40];

    const int tid  = threadIdx.x;
    const int wave = tid >> 5;
    const int lane = tid & 31;
    const int lr   = lane & 15;
    const int lh   = lane >> 4;
    const int wm   = wave >> 2;   // 0..1 -> 64 rows each
    const int wn   = wave & 3;    // 0..3 -> 32 cols each
    const int rowBase = blockIdx.y * BM;
    const int colBase = blockIdx.x * BN;

    // This thread's two 16B staging chunks (chunk c: row c>>2, halves (c&3)*8)
    const int c0r = tid >> 2,          c0o = (tid & 3) << 3;
    const int c1r = (tid + 256) >> 2,  c1o = c0o;

    v8f zero = {0.f,0.f,0.f,0.f,0.f,0.f,0.f,0.f};
    v8f acc[4][2];
    for (int mt = 0; mt < 4; ++mt)
        for (int nt = 0; nt < 2; ++nt) acc[mt][nt] = zero;

    // Async-copy one 128x32 K-tile of A and W into LDS buffer `buf`.
    auto issue_tile = [&](int buf, int k0) {
        // chunk 0
        {
            unsigned la = (unsigned)(unsigned long long)(uintptr_t)&As[buf][c0r][c0o];
            const _Float16* ga = &A[(size_t)(rowBase + c0r) * K + k0 + c0o];
            asm volatile("global_load_async_to_lds_b128 %0, %1, off"
                         :: "v"(la), "v"(ga) : "memory");
            unsigned lb = (unsigned)(unsigned long long)(uintptr_t)&Bs[buf][c0r][c0o];
            const _Float16* gb = &W[(size_t)(colBase + c0r) * K + k0 + c0o];
            asm volatile("global_load_async_to_lds_b128 %0, %1, off"
                         :: "v"(lb), "v"(gb) : "memory");
        }
        // chunk 1
        {
            unsigned la = (unsigned)(unsigned long long)(uintptr_t)&As[buf][c1r][c1o];
            const _Float16* ga = &A[(size_t)(rowBase + c1r) * K + k0 + c1o];
            asm volatile("global_load_async_to_lds_b128 %0, %1, off"
                         :: "v"(la), "v"(ga) : "memory");
            unsigned lb = (unsigned)(unsigned long long)(uintptr_t)&Bs[buf][c1r][c1o];
            const _Float16* gb = &W[(size_t)(colBase + c1r) * K + k0 + c1o];
            asm volatile("global_load_async_to_lds_b128 %0, %1, off"
                         :: "v"(lb), "v"(gb) : "memory");
        }
    };

    // Prologue: fill buffer 0 with K-tile 0.
    issue_tile(0, 0);
    asm volatile("s_wait_asynccnt 0" ::: "memory");
    __syncthreads();

    int buf = 0;
    for (int k0 = 0; k0 < K; k0 += BK) {
        if (k0 + BK < K) {
            issue_tile(buf ^ 1, k0 + BK);          // overlap next tile with compute
            if (k0 + 2 * BK < K) {                 // and prefetch the tile after
                __builtin_prefetch(&A[(size_t)(rowBase + c0r) * K + k0 + 2 * BK + c0o], 0, 3);
                __builtin_prefetch(&W[(size_t)(colBase + c0r) * K + k0 + 2 * BK + c0o], 0, 3);
            }
        }

        AFrag af[4]; BFrag bf[2];
        for (int mt = 0; mt < 4; ++mt) {
            int r = wm * 64 + mt * 16 + lr;
            #pragma unroll
            for (int j = 0; j < 8; ++j) {
                int kp = ((j >> 2) << 4) + (lh << 3) + ((j & 3) << 1); // A 16x32 layout
                af[mt].u[j] = *(const unsigned*)&As[buf][r][kp];
            }
        }
        for (int nt = 0; nt < 2; ++nt) {
            int r = wn * 32 + nt * 16 + lr;
            #pragma unroll
            for (int j = 0; j < 8; ++j) {
                int kp = ((j >> 2) << 4) + (lh << 3) + ((j & 3) << 1); // B 32x16 layout
                bf[nt].u[j] = *(const unsigned*)&Bs[buf][r][kp];
            }
        }
        #pragma unroll
        for (int mt = 0; mt < 4; ++mt)
            #pragma unroll
            for (int nt = 0; nt < 2; ++nt)
                acc[mt][nt] = __builtin_amdgcn_wmma_f32_16x16x32_f16(
                    false, af[mt].v, false, bf[nt].v, (short)0, acc[mt][nt], false, false);

        asm volatile("s_wait_asynccnt 0" ::: "memory"); // next tile landed in LDS
        __syncthreads();                                // all waves done with buf
        buf ^= 1;
    }

    // Epilogue: D layout -> VGPR j holds (M = j + lh*8, N = lr)
    for (int nt = 0; nt < 2; ++nt) {
        int col = colBase + wn * 32 + nt * 16 + lr;
        float bv = bias[col];
        for (int mt = 0; mt < 4; ++mt) {
            CFrag r; r.v = acc[mt][nt];
            #pragma unroll
            for (int j = 0; j < 8; ++j) {
                int row = rowBase + wm * 64 + mt * 16 + j + lh * 8;
                float val = r.f[j] + bv;
                size_t off = (size_t)row * Nout + col;
                if (accFlag) val += outF32[off];
                if (outF32) outF32[off] = val;
                if (outF16) outF16[off] = (_Float16)val;
            }
        }
    }
}

// ---------------------------------------------------------------------------
// Attention: one wave32 per token t = b*N + n. qkv row = 3072 f16:
// per head h (16): q at h*192, k at h*192+64, v at h*192+128 (d=0..63).
// scores(16x16) = q(16x64) @ k^T(64x16)  -> 2x wmma 16x16x32
// softmax over heads-g axis (16 lanes per half) via shfl_xor
// out(16x64) = attn(16x16, zero-padded K) @ v  -> 4x wmma 16x16x32
// Block: 256 threads = 8 waves, grid 1024.
// ---------------------------------------------------------------------------
__global__ __launch_bounds__(256) void attn_kernel(
    const _Float16* __restrict__ qkv, const float* __restrict__ pos_bias,
    _Float16* __restrict__ attn_out)
{
    __shared__ _Float16 att[8][16][16];

    const int tid  = threadIdx.x;
    const int wave = tid >> 5;
    const int lane = tid & 31;
    const int lr   = lane & 15;
    const int lh   = lane >> 4;
    const int t    = blockIdx.x * 8 + wave;   // 0..8191
    const int n    = t & 2047;                // seq index (N=2048)
    const _Float16* row = qkv + (size_t)t * 3072;

    // --- scores = q @ k^T ---
    AFrag a0, a1; BFrag b0, b1;
    #pragma unroll
    for (int j = 0; j < 8; ++j) {
        int kp = ((j >> 2) << 4) + (lh << 3) + ((j & 3) << 1);  // A: row=lr(head), k=d
        a0.u[j] = *(const unsigned*)(row + lr * 192 + kp);
        a1.u[j] = *(const unsigned*)(row + lr * 192 + 32 + kp);
        int kb = ((j >> 2) << 4) + (lh << 3) + ((j & 3) << 1);  // B: col=lr(g-head), k=d
        b0.u[j] = *(const unsigned*)(row + lr * 192 + 64 + kb);
        b1.u[j] = *(const unsigned*)(row + lr * 192 + 96 + kb);
    }
    v8f sc = {0.f,0.f,0.f,0.f,0.f,0.f,0.f,0.f};
    sc = __builtin_amdgcn_wmma_f32_16x16x32_f16(false, a0.v, false, b0.v, (short)0, sc, false, false);
    sc = __builtin_amdgcn_wmma_f32_16x16x32_f16(false, a1.v, false, b1.v, (short)0, sc, false, false);

    // --- scale + pos_bias + softmax over g (lanes within each 16-lane half) ---
    CFrag s; s.v = sc;
    const float scale = 0.125f;  // 64^-0.5
    #pragma unroll
    for (int j = 0; j < 8; ++j) {
        int h = j + lh * 8;
        float val = s.f[j] * scale + pos_bias[((size_t)n * 16 + h) * 16 + lr];
        float m = val;
        m = fmaxf(m, __shfl_xor(m, 1));
        m = fmaxf(m, __shfl_xor(m, 2));
        m = fmaxf(m, __shfl_xor(m, 4));
        m = fmaxf(m, __shfl_xor(m, 8));
        float e = __expf(val - m);
        float su = e;
        su += __shfl_xor(su, 1);
        su += __shfl_xor(su, 2);
        su += __shfl_xor(su, 4);
        su += __shfl_xor(su, 8);
        att[wave][h][lr] = (_Float16)(e / su);   // plain [h][g] tile in LDS
    }
    asm volatile("s_wait_dscnt 0" ::: "memory"); // same-wave LDS RAW, CDNA5 split counter

    // --- transpose-read attn into A layout (row=lr, k=g; g<16 valid, rest 0) ---
    AFrag aa;
    #pragma unroll
    for (int j = 0; j < 4; ++j)
        aa.u[j] = *(const unsigned*)&att[wave][lr][(lh << 3) + (j << 1)];
    #pragma unroll
    for (int j = 4; j < 8; ++j) aa.u[j] = 0u;

    // --- out = attn @ v, 4 chunks of 16 d-columns ---
    const _Float16* vrow = row + 128;
    for (int c = 0; c < 4; ++c) {
        BFrag bv;
        #pragma unroll
        for (int j = 0; j < 8; ++j) {
            _Float16 lo = (_Float16)0.f, hi = (_Float16)0.f;
            if (j < 4) {                         // k=g=8*lh+2j(+1) in 0..15; j>=4 carry pad zeros
                lo = vrow[(size_t)((lh << 3) + 2 * j)     * 192 + c * 16 + lr];
                hi = vrow[(size_t)((lh << 3) + 2 * j + 1) * 192 + c * 16 + lr];
            }
            bv.h[2 * j]     = lo;
            bv.h[2 * j + 1] = hi;
        }
        v8f o = {0.f,0.f,0.f,0.f,0.f,0.f,0.f,0.f};
        o = __builtin_amdgcn_wmma_f32_16x16x32_f16(false, aa.v, false, bv.v, (short)0, o, false, false);
        CFrag of; of.v = o;
        #pragma unroll
        for (int j = 0; j < 8; ++j) {
            int h = j + lh * 8;
            attn_out[(size_t)t * 1024 + h * 64 + c * 16 + lr] = (_Float16)of.f[j];
        }
    }
}

// ---------------------------------------------------------------------------
// Depthwise conv k=3, pad=1, per channel (f16 in/out, f32 math)
// ---------------------------------------------------------------------------
__global__ __launch_bounds__(256) void dwconv_kernel(
    const _Float16* __restrict__ in, const float* __restrict__ Wdw,
    const float* __restrict__ bdw, _Float16* __restrict__ out)
{
    int idx = blockIdx.x * 256 + threadIdx.x;   // over B*N*C = 8388608
    int c = idx & 1023;
    int nn = (idx >> 10) & 2047;
    float acc = bdw[c] + (float)in[idx] * Wdw[c * 3 + 1];
    if (nn > 0)    acc += (float)in[idx - 1024] * Wdw[c * 3 + 0];
    if (nn < 2047) acc += (float)in[idx + 1024] * Wdw[c * 3 + 2];
    out[idx] = (_Float16)acc;
}

// ---------------------------------------------------------------------------
extern "C" void kernel_launch(void* const* d_in, const int* in_sizes, int n_in,
                              void* d_out, int out_size, void* d_ws, size_t ws_size,
                              hipStream_t stream) {
    const float* x    = (const float*)d_in[0];
    const float* Wqkv = (const float*)d_in[1];
    const float* bqkv = (const float*)d_in[2];
    const float* Wout = (const float*)d_in[3];
    const float* bout = (const float*)d_in[4];
    const float* pb   = (const float*)d_in[5];
    const float* Wcp  = (const float*)d_in[6];
    const float* bcp  = (const float*)d_in[7];
    const float* Wpw  = (const float*)d_in[8];
    const float* bpw  = (const float*)d_in[9];
    const float* Wdw  = (const float*)d_in[10];
    const float* bdw  = (const float*)d_in[11];
    const float* Wtw  = (const float*)d_in[12];
    const float* btw  = (const float*)d_in[13];
    float* out = (float*)d_out;

    const size_t BNC  = 8192u * 1024u;   // 8,388,608
    const size_t QKVN = 8192u * 3072u;   // 25,165,824
    const size_t WQKV = 3072u * 1024u;
    const size_t WSQ  = 1024u * 1024u;

    char* ws = (char*)d_ws;
    size_t off = 0;
    auto alloc = [&](size_t elems) {
        _Float16* p = (_Float16*)(ws + off);
        off += ((elems * sizeof(_Float16) + 255) / 256) * 256;
        return p;
    };
    _Float16* x_h    = alloc(BNC);
    _Float16* Wqkv_h = alloc(WQKV);
    _Float16* Wout_h = alloc(WSQ);
    _Float16* Wcp_h  = alloc(WSQ);
    _Float16* Wpw_h  = alloc(WSQ);
    _Float16* Wtw_h  = alloc(WSQ);
    _Float16* qkv_h  = alloc(QKVN);
    _Float16* attn_h = alloc(BNC);
    _Float16* xc1_h  = alloc(BNC);
    _Float16* xc2_h  = alloc(BNC);
    _Float16* xd_h   = xc1_h;  // alias: xc1 dead after pointwise GEMM

    dim3 blk(256);
    auto cvt = [&](const float* src, _Float16* dst, size_t n) {
        cvt_f32_f16<<<dim3((unsigned)((n + 255) / 256)), blk, 0, stream>>>(src, dst, (int)n);
    };
    cvt(x,    x_h,    BNC);
    cvt(Wqkv, Wqkv_h, WQKV);
    cvt(Wout, Wout_h, WSQ);
    cvt(Wcp,  Wcp_h,  WSQ);
    cvt(Wpw,  Wpw_h,  WSQ);
    cvt(Wtw,  Wtw_h,  WSQ);

    // Convolutional branch -> writes d_out (f32)
    gemm_f16<<<dim3(8, 64),  blk, 0, stream>>>(x_h,   Wcp_h, bcp, nullptr, xc1_h, 1024, 1024, 0);
    gemm_f16<<<dim3(8, 64),  blk, 0, stream>>>(xc1_h, Wpw_h, bpw, nullptr, xc2_h, 1024, 1024, 0);
    dwconv_kernel<<<dim3(32768), blk, 0, stream>>>(xc2_h, Wdw, bdw, xd_h);
    gemm_f16<<<dim3(8, 64),  blk, 0, stream>>>(xd_h,  Wtw_h, btw, out, nullptr, 1024, 1024, 0);

    // Attention branch -> accumulates into d_out
    gemm_f16<<<dim3(24, 64), blk, 0, stream>>>(x_h, Wqkv_h, bqkv, nullptr, qkv_h, 3072, 1024, 0);
    attn_kernel<<<dim3(1024), blk, 0, stream>>>(qkv_h, pb, attn_h);
    gemm_f16<<<dim3(8, 64),  blk, 0, stream>>>(attn_h, Wout_h, bout, out, nullptr, 1024, 1024, 1);
}
